// AttnBlock_41841571397721
// MI455X (gfx1250) — matrix operand order repaired and checked by reading, not compiled, not run on hardware
//
#include <hip/hip_runtime.h>

// ---------------------------------------------------------------------------
// AttnBlock for MI455X (gfx1250): GroupNorm + QKV + softmax(QK^T)V + proj + res
// All GEMMs via v_wmma_f32_16x16x32_bf16 (wave32, f32 accum).
// ---------------------------------------------------------------------------

typedef __bf16 v8bf  __attribute__((ext_vector_type(8)));
typedef __bf16 v16bf __attribute__((ext_vector_type(16)));
typedef float  v8f   __attribute__((ext_vector_type(8)));

#define BATCH 16
#define CCH   512
#define NTOK  1024
#define GROUPS 32

__device__ __forceinline__ v16bf cat16(v8bf lo, v8bf hi) {
    return __builtin_shufflevector(lo, hi, 0,1,2,3,4,5,6,7,8,9,10,11,12,13,14,15);
}

__device__ __forceinline__ v8f wmma_bf16(v16bf a, v16bf b, v8f c) {
    return __builtin_amdgcn_wmma_f32_16x16x32_bf16(false, a, false, b, (short)0, c,
                                                   false, false);
}

// A-fragment: 16x32 tile from row-major [*, ld] matrix, rows base n0.
// lane<16 holds K {k0..k0+7, k0+16..k0+23}; lane>=16 holds {k0+8.., k0+24..}.
__device__ __forceinline__ v16bf load_afrag(const __bf16* base, int n0, int k0,
                                            int lr, int half, int ld) {
    const __bf16* rp = base + (size_t)(n0 + lr) * ld + k0 + 8 * half;
    v8bf lo = *(const v8bf*)rp;
    v8bf hi = *(const v8bf*)(rp + 16);
    return cat16(lo, hi);
}

// B-fragment: 32x16 tile; memory holds B^T row-major ([cols, K] layout).
// lane holds column (lane&15), K {k0+8h..+7, k0+16+8h..+7} (h = lane>>4).
__device__ __forceinline__ v16bf load_bfrag(const __bf16* bt, int col0, int k0,
                                            int lr, int half, int ld) {
    const __bf16* rp = bt + (size_t)(col0 + lr) * ld + k0 + 8 * half;
    return cat16(*(const v8bf*)rp, *(const v8bf*)(rp + 16));
}

// ------------------------------- kernel 0: weights f32 -> bf16 --------------
__global__ __launch_bounds__(256) void cvt_weights(const float* wq, const float* wk,
                                                   const float* wv, const float* wp,
                                                   __bf16* wbf) {
    int tid = blockIdx.x * 256 + threadIdx.x;          // 4 * 512*512
    int m = tid >> 18;
    int i = tid & ((1 << 18) - 1);
    const float* src = (m == 0) ? wq : (m == 1) ? wk : (m == 2) ? wv : wp;
    wbf[tid] = (__bf16)src[i];
}

// ------------------------------- kernel 1: GroupNorm stats ------------------
__global__ __launch_bounds__(256) void gn_stats(const float* x, float* stats) {
    int bg = blockIdx.x;                               // 512 = B*GROUPS
    const float* p = x + (size_t)bg * 16384;           // 16 ch * 1024 px contiguous
    float s = 0.f, s2 = 0.f;
    for (int i = threadIdx.x; i < 16384; i += 256) {
        float v = p[i];
        s += v; s2 += v * v;
    }
    __shared__ float sh[256], sh2[256];
    sh[threadIdx.x] = s; sh2[threadIdx.x] = s2;
    __syncthreads();
    for (int st = 128; st > 0; st >>= 1) {
        if (threadIdx.x < st) {
            sh[threadIdx.x] += sh[threadIdx.x + st];
            sh2[threadIdx.x] += sh2[threadIdx.x + st];
        }
        __syncthreads();
    }
    if (threadIdx.x == 0) {
        float mu = sh[0] * (1.f / 16384.f);
        float var = sh2[0] * (1.f / 16384.f) - mu * mu;
        stats[bg * 2] = mu;
        stats[bg * 2 + 1] = rsqrtf(var + 1e-5f);
    }
}

// -------------------- kernel 2: normalize + transpose to [B,N,C] bf16 -------
__global__ __launch_bounds__(256) void gn_apply(const float* x, const float* gw,
                                                const float* gb, const float* stats,
                                                __bf16* hn) {
    int tid = blockIdx.x * 256 + threadIdx.x;          // B*(C/8)*N = 1,048,576
    int n = tid & (NTOK - 1);
    int t = tid >> 10;
    int c0 = (t & 63) * 8;
    int b = t >> 6;
    const float* xp = x + ((size_t)b * CCH + c0) * NTOK + n;
    v8bf ov;
#pragma unroll
    for (int i = 0; i < 8; i++) {
        int c = c0 + i;
        int g = c >> 4;
        float mu = stats[(b * GROUPS + g) * 2];
        float rstd = stats[(b * GROUPS + g) * 2 + 1];
        float v = xp[(size_t)i * NTOK];
        ov[i] = (__bf16)(((v - mu) * rstd) * gw[c] + gb[c]);
    }
    *(v8bf*)(hn + ((size_t)b * NTOK + n) * CCH + c0) = ov;
}

// -------------------- kernel 3: QKV projection GEMMs ------------------------
// q,k stored [B,N,C]; v stored transposed [B,C,N].
__global__ __launch_bounds__(32) void qkv_gemm(const __bf16* hn, const __bf16* wbf,
                                               const float* bq, const float* bk,
                                               const float* bv, __bf16* q, __bf16* k,
                                               __bf16* vt) {
    int nt = blockIdx.x;                 // 64
    int og = blockIdx.y;                 // 8 (4 tiles of 16 each)
    int bz = blockIdx.z;                 // 48 = b*3 + mat
    int b = bz / 3, mat = bz % 3;
    int lane = threadIdx.x, half = lane >> 4, lr = lane & 15;
    int n0 = nt * 16;
    const __bf16* A = hn + (size_t)b * NTOK * CCH;
    const __bf16* W = wbf + (size_t)mat * CCH * CCH;     // [O,C] row-major == B^T
    const float* bias = (mat == 0) ? bq : (mat == 1) ? bk : bv;

    v8f acc[4] = {v8f{}, v8f{}, v8f{}, v8f{}};
#pragma unroll
    for (int kk = 0; kk < 16; kk++) {
        int k0 = kk * 32;
        v16bf a = load_afrag(A, n0, k0, lr, half, CCH);
#pragma unroll
        for (int j = 0; j < 4; j++) {
            v16bf bf = load_bfrag(W, og * 64 + j * 16, k0, lr, half, CCH);
            acc[j] = wmma_bf16(a, bf, acc[j]);
        }
    }
#pragma unroll
    for (int j = 0; j < 4; j++) {
        int o0 = og * 64 + j * 16;
        float bia = bias[o0 + lr];
        if (mat < 2) {
            __bf16* dst = ((mat == 0) ? q : k) + (size_t)b * NTOK * CCH;
#pragma unroll
            for (int r = 0; r < 8; r++) {
                int n = n0 + r + 8 * half;
                dst[(size_t)n * CCH + o0 + lr] = (__bf16)(acc[j][r] + bia);
            }
        } else {
            // vT[c][m]: contiguous 8 bf16 per lane
            v8bf pk;
#pragma unroll
            for (int r = 0; r < 8; r++) pk[r] = (__bf16)(acc[j][r] + bia);
            __bf16* dst = vt + (size_t)b * CCH * NTOK + (size_t)(o0 + lr) * NTOK +
                          n0 + 8 * half;
            *(v8bf*)dst = pk;
        }
    }
}

// -------------------- kernel 4: attention (scores + softmax + P@V) ----------
__global__ __launch_bounds__(32) void attn_core(const __bf16* qb, const __bf16* kb,
                                                const __bf16* vtb, __bf16* ob) {
    __shared__ __bf16 S[16 * NTOK];                    // 32 KB score rows
    __shared__ float rsum_s[16];
    int b = blockIdx.y;
    int n0 = blockIdx.x * 16;
    int lane = threadIdx.x, half = lane >> 4, lr = lane & 15;
    const __bf16* q = qb + (size_t)b * NTOK * CCH;
    const __bf16* kmat = kb + (size_t)b * NTOK * CCH;
    const __bf16* vt = vtb + (size_t)b * CCH * NTOK;

    // preload the 16 Q A-fragments for this row tile (K = 512); kept in VGPRs
    v16bf qa[16];
#pragma unroll
    for (int kk = 0; kk < 16; kk++)
        qa[kk] = load_afrag(q, n0, kk * 32, lr, half, CCH);

    const float scale = 0.04419417382415922f;          // 512^-0.5
    for (int mt = 0; mt < 64; mt++) {
        v8f acc = v8f{};
        int m0 = mt * 16;
#pragma unroll
        for (int kk = 0; kk < 16; kk++) {
            v16bf bf = load_bfrag(kmat, m0, kk * 32, lr, half, CCH);
            acc = wmma_bf16(qa[kk], bf, acc);
        }
#pragma unroll
        for (int r = 0; r < 8; r++) {
            int row = r + 8 * half;
            S[row * NTOK + m0 + lr] = (__bf16)(acc[r] * scale);
        }
    }
    __syncthreads();

    // softmax: lane handles row (lane&15), column half 512*(lane>>4).
    // Vectorized 16B LDS traffic (ds_load_b128 / ds_store_b128).
    __bf16* rowp = &S[lr * NTOK + half * 512];
    float mx = -1e30f;
    for (int i = 0; i < 512; i += 8) {
        v8bf v = *(const v8bf*)(rowp + i);
#pragma unroll
        for (int j = 0; j < 8; j++) mx = fmaxf(mx, (float)v[j]);
    }
    mx = fmaxf(mx, __shfl_xor(mx, 16, 32));
    float sum = 0.f;
    for (int i = 0; i < 512; i += 8) {
        v8bf v = *(const v8bf*)(rowp + i);
        v8bf pv;
#pragma unroll
        for (int j = 0; j < 8; j++) {
            float p = __expf((float)v[j] - mx);
            sum += p;
            pv[j] = (__bf16)p;
        }
        *(v8bf*)(rowp + i) = pv;
    }
    sum += __shfl_xor(sum, 16, 32);
    if (half == 0) rsum_s[lr] = 1.0f / sum;
    __syncthreads();
    float rs[8];
#pragma unroll
    for (int r = 0; r < 8; r++) rs[r] = rsum_s[r + 8 * half];

    // O = P @ V, one 16-col tile of C at a time; A from LDS, B from vT
    for (int ct = 0; ct < 32; ct++) {
        v8f acc = v8f{};
        for (int mk = 0; mk < 32; mk++) {
            int m0 = mk * 32;
            const __bf16* ap = &S[lr * NTOK + m0 + 8 * half];
            v16bf a = cat16(*(const v8bf*)ap, *(const v8bf*)(ap + 16));
            v16bf bf = load_bfrag(vt, ct * 16, m0, lr, half, NTOK);
            acc = wmma_bf16(a, bf, acc);
        }
#pragma unroll
        for (int r = 0; r < 8; r++) {
            int n = n0 + r + 8 * half;
            ob[(size_t)b * NTOK * CCH + (size_t)n * CCH + ct * 16 + lr] =
                (__bf16)(acc[r] * rs[r]);
        }
    }
}

// -------------------- kernel 5: out projection + bias + residual ------------
__global__ __launch_bounds__(32) void out_proj(const __bf16* ob, const __bf16* wpbf,
                                               const float* bp, const float* x,
                                               float* out) {
    int nt = blockIdx.x;                 // 64
    int og = blockIdx.y;                 // 8
    int b = blockIdx.z;                  // 16
    int lane = threadIdx.x, half = lane >> 4, lr = lane & 15;
    int n0 = nt * 16;
    const __bf16* A = ob + (size_t)b * NTOK * CCH;

    v8f acc[4] = {v8f{}, v8f{}, v8f{}, v8f{}};
#pragma unroll
    for (int kk = 0; kk < 16; kk++) {
        int k0 = kk * 32;
        v16bf a = load_afrag(A, n0, k0, lr, half, CCH);
#pragma unroll
        for (int j = 0; j < 4; j++) {
            v16bf bf = load_bfrag(wpbf, og * 64 + j * 16, k0, lr, half, CCH);
            acc[j] = wmma_bf16(a, bf, acc[j]);
        }
    }
#pragma unroll
    for (int j = 0; j < 4; j++) {
        int o0 = og * 64 + j * 16;
        float bia = bp[o0 + lr];
        // out layout [B,C,N]: lane's 8 regs are 8 consecutive n -> 32B store
        size_t base = (size_t)b * CCH * NTOK + (size_t)(o0 + lr) * NTOK + n0 + 8 * half;
        v8f xv = *(const v8f*)(x + base);
        v8f rv;
#pragma unroll
        for (int r = 0; r < 8; r++) rv[r] = xv[r] + acc[j][r] + bia;
        *(v8f*)(out + base) = rv;
    }
}

// ---------------------------------------------------------------------------
extern "C" void kernel_launch(void* const* d_in, const int* in_sizes, int n_in,
                              void* d_out, int out_size, void* d_ws, size_t ws_size,
                              hipStream_t stream) {
    const float* x   = (const float*)d_in[0];
    const float* gnw = (const float*)d_in[1];
    const float* gnb = (const float*)d_in[2];
    const float* wq  = (const float*)d_in[3];
    const float* bq  = (const float*)d_in[4];
    const float* wk  = (const float*)d_in[5];
    const float* bk  = (const float*)d_in[6];
    const float* wv  = (const float*)d_in[7];
    const float* bv  = (const float*)d_in[8];
    const float* wp  = (const float*)d_in[9];
    const float* bp  = (const float*)d_in[10];
    float* out = (float*)d_out;

    char* ws = (char*)d_ws;
    const size_t MAT = (size_t)BATCH * NTOK * CCH * sizeof(__bf16);   // 16 MB
    float*  stats = (float*)(ws);                                     // 4 KB
    __bf16* wbf   = (__bf16*)(ws + 4096);                             // 2 MB (q,k,v,p)
    __bf16* hn    = (__bf16*)(ws + 4096 + 4 * (size_t)CCH * CCH * 2);
    __bf16* qbuf  = (__bf16*)((char*)hn + MAT);
    __bf16* kbuf  = (__bf16*)((char*)qbuf + MAT);
    __bf16* vtbuf = (__bf16*)((char*)kbuf + MAT);
    __bf16* obuf  = hn;  // hn dead after qkv_gemm; reuse for attention output

    cvt_weights<<<4096, 256, 0, stream>>>(wq, wk, wv, wp, wbf);
    gn_stats<<<BATCH * GROUPS, 256, 0, stream>>>(x, stats);
    gn_apply<<<4096, 256, 0, stream>>>(x, gnw, gnb, stats, hn);
    qkv_gemm<<<dim3(64, 8, BATCH * 3), 32, 0, stream>>>(hn, wbf, bq, bk, bv,
                                                        qbuf, kbuf, vtbuf);
    attn_core<<<dim3(64, BATCH), 32, 0, stream>>>(qbuf, kbuf, vtbuf, obuf);
    out_proj<<<dim3(64, 8, BATCH), 32, 0, stream>>>(
        obuf, wbf + 3 * (size_t)CCH * CCH, bp, x, out);
}
